// drugcell_nn_32873679684080
// MI455X (gfx1250) — matrix operand order repaired and checked by reading, not compiled, not run on hardware
//
#include <hip/hip_runtime.h>
#include <hip/hip_bf16.h>

// ---------------------------------------------------------------------------
// DrugCell-style hierarchical NN for MI455X (gfx1250, wave32, WMMA).
//
//  * Fold Wf@Wg (no nonlinearity between einsum1 and einsum2) -> one
//    v_wmma_f32_16x16x32_f16 per 16-row batch tile (K=2G=32 exact fit).
//  * One block per term; full 2048-row activation tile kept in 48KB LDS so
//    the two-pass batch-norm never touches HBM.
//  * zg stored f16 in workspace (only consumed as WMMA A operand),
//    hier outputs stored f32 (root feeds the f32 softmax head).
//  * Branch-free tanh (1 trans + ~5 VALU, no exec-mask divergence).
//  * A-fragment K-map per ISA A-table; B-fragment K-map per ISA B-layout
//    (lanes 0-15: K={0..7,16..23}, lanes 16-31: K={8..15,24..31}, col = lane&15).
// ---------------------------------------------------------------------------

#define TTOT   1365
#define BATCH  2048
#define NGENE_ 3008
#define GG     16
#define HH     6
#define NCLS   30
#define OUTW   (NCLS + TTOT)   // 1395
#define EPS_   1e-5f

typedef __attribute__((ext_vector_type(16))) _Float16 v16h;
typedef __attribute__((ext_vector_type(8)))  float    v8f;

union AFrag { v16h v; _Float16 h[16]; };
union H2   { _Float16 h[2]; unsigned u; };

// A-matrix (16x32 f16) element->K: lanes 0-15: {0..7,16..23}; 16-31: {8..15,24..31}
__device__ __forceinline__ int kA(int e, int hi) {
  return (hi ? 8 : 0) + e + ((e < 8) ? 0 : 8);
}
// B-matrix (32x16 f16) element->K: lanes 0-15: {0..7,16..23}; 16-31: {8..15,24..31}
__device__ __forceinline__ int kB(int e, int hi) {
  return (hi ? 8 : 0) + e + ((e < 8) ? 0 : 8);
}

// Branch-free tanh: copysign(1 - 2/(exp(2|x|)+1), x).
// Saturates cleanly (rcp(inf)=0); abs error ~1e-7 near zero.
__device__ __forceinline__ float fast_tanh(float v) {
  float ax = __builtin_fabsf(v);
  float e  = __expf(2.f * ax);
  float r  = 1.f - 2.f * __builtin_amdgcn_rcpf(e + 1.f);
  return __builtin_copysignf(r, v);
}

// ---------------------------------------------------------------------------
// Kernel 1: per-term gather + fused GEMM (feat @ (Wf@Wg)) + tanh + batchnorm.
// grid = T terms, block = 256 (8 wave32).
// ---------------------------------------------------------------------------
__global__ __launch_bounds__(256) void term_forward(
    const float* __restrict__ x, const float* __restrict__ y,
    const int*   __restrict__ gene_idx,
    const float* __restrict__ Wf, const float* __restrict__ bf,
    const float* __restrict__ Wg, const float* __restrict__ bg,
    const float* __restrict__ gamma, const float* __restrict__ beta,
    _Float16* __restrict__ zg)
{
  const int t    = blockIdx.x;
  const int tid  = threadIdx.x;
  const int lane = tid & 31;
  const int wave = tid >> 5;
  const int hi   = lane >> 4;
  const int n    = lane & 15;       // A row within tile AND B/C column

  __shared__ float lds_z[BATCH * HH];     // 48 KB pre-BN activations
  __shared__ float weff[2 * GG * HH];     // Wf@Wg  (32 x 6)
  __shared__ float beff[HH];              // bf@Wg + bg
  __shared__ int   gidx[GG];
  __shared__ float psum[256], pssq[256];
  __shared__ float sc[HH], sh[HH];

  if (tid < GG) gidx[tid] = gene_idx[t * GG + tid];
  if (tid < 2 * GG * HH) {                // 192 threads: Weff[g][h]
    int g = tid / HH, h = tid % HH;
    float s = 0.f;
    for (int f = 0; f < HH; ++f)
      s += Wf[(t * 2 * GG + g) * HH + f] * Wg[(t * HH + f) * HH + h];
    weff[tid] = s;
  }
  if (tid >= 192 && tid < 192 + HH) {     // beff[h]
    int h = tid - 192;
    float s = bg[t * HH + h];
    for (int f = 0; f < HH; ++f) s += bf[t * HH + f] * Wg[(t * HH + f) * HH + h];
    beff[h] = s;
  }
  __syncthreads();

  // B fragment: K=32 x N=16, columns >= 6 are zero padding.
  AFrag bfr;
#pragma unroll
  for (int e = 0; e < 16; ++e) {
    int k = kB(e, hi);
    bfr.h[e] = (n < HH) ? (_Float16)weff[k * HH + n] : (_Float16)0.f;
  }
  const float bb = (n < HH) ? beff[n] : 0.f;

  float acc = 0.f, accq = 0.f;

  for (int tile = 0; tile < 16; ++tile) { // 8 waves x 16 tiles x 16 rows = 2048
    const int base = wave * 256 + tile * 16;
    const int b    = base + n;            // batch row this lane gathers
    AFrag a;
#pragma unroll
    for (int e = 0; e < 16; ++e) {
      int k = kA(e, hi);
      int g = gidx[k & 15];
      float fv = (k < GG) ? x[b * NGENE_ + g] : y[b * NGENE_ + g];
      a.h[e] = (_Float16)fv;
    }
    v8f c = {};
    c = __builtin_amdgcn_wmma_f32_16x16x32_f16(false, a.v, false, bfr.v,
                                               (short)0, c, false, false);
    if (n < HH) {
#pragma unroll
      for (int vv = 0; vv < 8; ++vv) {
        int r = base + vv + (hi ? 8 : 0);
        float z = fast_tanh(c[vv] + bb);
        lds_z[r * HH + n] = z;
        acc  += z;
        accq += z * z;
      }
    }
  }

  psum[tid] = acc; pssq[tid] = accq;
  __syncthreads();
  if (tid < HH) {                         // reduce 16 partials per channel
    float s = 0.f, q = 0.f;
    for (int w = 0; w < 8; ++w) {
      s += psum[w * 32 + tid] + psum[w * 32 + 16 + tid];
      q += pssq[w * 32 + tid] + pssq[w * 32 + 16 + tid];
    }
    float m   = s / (float)BATCH;
    float var = q / (float)BATCH - m * m;
    float scl = gamma[t * HH + tid] * rsqrtf(var + EPS_);
    sc[tid] = scl;
    sh[tid] = beta[t * HH + tid] - m * scl;
  }
  __syncthreads();

  // Apply BN, pack 6 halfs -> 3 dwords per row (coalesced 12B/row writes).
  for (int b = tid; b < BATCH; b += 256) {
    const float* zr = &lds_z[b * HH];
    unsigned* zp = (unsigned*)(zg + (size_t)(t * BATCH + b) * HH);
#pragma unroll
    for (int j = 0; j < 3; ++j) {
      H2 p;
      p.h[0] = (_Float16)(zr[2 * j]     * sc[2 * j]     + sh[2 * j]);
      p.h[1] = (_Float16)(zr[2 * j + 1] * sc[2 * j + 1] + sh[2 * j + 1]);
      zp[j] = p.u;
    }
  }
}

// ---------------------------------------------------------------------------
// Kernel 2: one hierarchy layer. grid = Tl, block = 256.
// inp = [children (cdim=24, prev layer f32) | direct (6, zg f16)], K pad 32.
// v_wmma per 16-row tile; tanh + full-batch BN + aux head. cdim==0 layer 0.
// ---------------------------------------------------------------------------
__global__ __launch_bounds__(256) void hier_layer(
    const _Float16* __restrict__ zg, float* __restrict__ hout,
    const float* __restrict__ W,     const float* __restrict__ bias,
    const float* __restrict__ gamma, const float* __restrict__ beta,
    const float* __restrict__ a1w,   const float* __restrict__ a1b,
    const float* __restrict__ a2w,   const float* __restrict__ a2b,
    float* __restrict__ out, int goff, int goffp, int cdim)
{
  const int t    = blockIdx.x;
  const int gt   = goff + t;
  const int din  = cdim + HH;
  const int tid  = threadIdx.x;
  const int lane = tid & 31;
  const int wave = tid >> 5;
  const int hi   = lane >> 4;
  const int n    = lane & 15;

  __shared__ float lds_t[BATCH * HH];     // 48 KB
  __shared__ float psum[256], pssq[256];
  __shared__ float sc[HH], shf[HH];

  AFrag bfr;                              // W: din x 6, pad K->32, N->16
#pragma unroll
  for (int e = 0; e < 16; ++e) {
    int k = kB(e, hi);
    bfr.h[e] = (n < HH && k < din) ? (_Float16)W[(t * din + k) * HH + n]
                                   : (_Float16)0.f;
  }
  const float bb = (n < HH) ? bias[t * HH + n] : 0.f;

  float acc = 0.f, accq = 0.f;

  for (int tile = 0; tile < 16; ++tile) {
    const int base = wave * 256 + tile * 16;
    const int b    = base + n;
    AFrag a;
#pragma unroll
    for (int e = 0; e < 16; ++e) {
      int k = kA(e, hi);
      if (k < cdim) {                     // child: prev layer, term 4t + k/6
        int ct = goffp + t * 4 + k / HH;
        a.h[e] = (_Float16)hout[(ct * BATCH + b) * HH + (k % HH)];
      } else if (k < din) {               // direct: zg, already f16
        a.h[e] = zg[(gt * BATCH + b) * HH + (k - cdim)];
      } else {
        a.h[e] = (_Float16)0.f;
      }
    }
    v8f c = {};
    c = __builtin_amdgcn_wmma_f32_16x16x32_f16(false, a.v, false, bfr.v,
                                               (short)0, c, false, false);
    if (n < HH) {
#pragma unroll
      for (int vv = 0; vv < 8; ++vv) {
        int r = base + vv + (hi ? 8 : 0);
        float tv = fast_tanh(c[vv] + bb);
        lds_t[r * HH + n] = tv;
        acc  += tv;
        accq += tv * tv;
      }
    }
  }

  psum[tid] = acc; pssq[tid] = accq;
  __syncthreads();
  if (tid < HH) {
    float s = 0.f, q = 0.f;
    for (int w = 0; w < 8; ++w) {
      s += psum[w * 32 + tid] + psum[w * 32 + 16 + tid];
      q += pssq[w * 32 + tid] + pssq[w * 32 + 16 + tid];
    }
    float m   = s / (float)BATCH;
    float var = q / (float)BATCH - m * m;
    float scl = gamma[t * HH + tid] * rsqrtf(var + EPS_);
    sc[tid]  = scl;
    shf[tid] = beta[t * HH + tid] - m * scl;
  }
  __syncthreads();

  const float w2 = a2w[t], b2 = a2b[t], b1 = a1b[t];
  float w1[HH];
#pragma unroll
  for (int h = 0; h < HH; ++h) w1[h] = a1w[t * HH + h];

  for (int b = tid; b < BATCH; b += 256) {
    float o[HH];
    float s1 = b1;
#pragma unroll
    for (int h = 0; h < HH; ++h) {
      o[h] = lds_t[b * HH + h] * sc[h] + shf[h];
      s1 += o[h] * w1[h];
    }
    float2* hp = (float2*)&hout[(size_t)(gt * BATCH + b) * HH]; // 24B rows, 8-aligned
    hp[0] = make_float2(o[0], o[1]);
    hp[1] = make_float2(o[2], o[3]);
    hp[2] = make_float2(o[4], o[5]);
    out[b * OUTW + NCLS + gt] = fast_tanh(s1) * w2 + b2;  // aux output column
  }
}

// ---------------------------------------------------------------------------
// Kernel 3: final softmax head on root term: (B x 6) @ (6 x 30).
// ---------------------------------------------------------------------------
__global__ __launch_bounds__(256) void final_head(
    const float* __restrict__ hout, const float* __restrict__ Wfin,
    const float* __restrict__ bfin, float* __restrict__ out, int rootGt)
{
  __shared__ float w[HH * NCLS];
  __shared__ float bbv[NCLS];
  const int tid = threadIdx.x;
  if (tid < HH * NCLS) w[tid] = Wfin[tid];
  if (tid < NCLS)      bbv[tid] = bfin[tid];
  __syncthreads();

  const int b = blockIdx.x * 256 + tid;
  if (b >= BATCH) return;

  float h[HH];
#pragma unroll
  for (int i = 0; i < HH; ++i) h[i] = hout[(rootGt * BATCH + b) * HH + i];

  float lg[NCLS];
  float mx = -3.4e38f;
#pragma unroll
  for (int c = 0; c < NCLS; ++c) {
    float s = bbv[c];
#pragma unroll
    for (int i = 0; i < HH; ++i) s += h[i] * w[i * NCLS + c];
    lg[c] = s;
    mx = fmaxf(mx, s);
  }
  float sum = 0.f;
#pragma unroll
  for (int c = 0; c < NCLS; ++c) { lg[c] = expf(lg[c] - mx); sum += lg[c]; }
  const float inv = 1.f / sum;
#pragma unroll
  for (int c = 0; c < NCLS; ++c) out[b * OUTW + c] = lg[c] * inv;
}

// ---------------------------------------------------------------------------
// Host launcher. Input order per setup_inputs() dict (tuples flatten):
//  0:x 1:y 2:gene_idx 3:Wf 4:bf 5:Wg 6:bg 7:g_gamma 8:g_beta 9:g_a1w
//  10:g_a1b 11:g_a2w 12:g_a2b 13:Wfin 14:bfin
//  15..20:hier_W[l] 21..26:hier_b[l] 27..32:hier_gamma[l] 33..38:hier_beta[l]
//  39..44:hier_a1w[l] 45..50:hier_a1b[l] 51..56:hier_a2w[l] 57..62:hier_a2b[l]
// (g_a1w/g_a1b/g_a2w/g_a2b are dead code in the reference -> unused.)
// ---------------------------------------------------------------------------
extern "C" void kernel_launch(void* const* d_in, const int* in_sizes, int n_in,
                              void* d_out, int out_size, void* d_ws, size_t ws_size,
                              hipStream_t stream) {
  (void)in_sizes; (void)n_in; (void)out_size; (void)ws_size;

  const float* x    = (const float*)d_in[0];
  const float* y    = (const float*)d_in[1];
  const int*   gi   = (const int*)  d_in[2];
  const float* Wf   = (const float*)d_in[3];
  const float* bf   = (const float*)d_in[4];
  const float* Wg   = (const float*)d_in[5];
  const float* bg   = (const float*)d_in[6];
  const float* gga  = (const float*)d_in[7];
  const float* gbe  = (const float*)d_in[8];
  const float* Wfin = (const float*)d_in[13];
  const float* bfin = (const float*)d_in[14];
  float* out = (float*)d_out;

  // Workspace: zg as f16 (T*B*6 = 33.5 MB), hier outputs f32 (67 MB).
  _Float16* zg = (_Float16*)d_ws;
  size_t zgBytes = (size_t)TTOT * BATCH * HH * sizeof(_Float16);
  float* hout = (float*)((char*)d_ws + ((zgBytes + 255) & ~(size_t)255));

  term_forward<<<TTOT, 256, 0, stream>>>(x, y, gi, Wf, bf, Wg, bg, gga, gbe, zg);

  const int LAY[6] = {1024, 256, 64, 16, 4, 1};
  int off[6]; off[0] = 0;
  for (int l = 1; l < 6; ++l) off[l] = off[l - 1] + LAY[l - 1];

  for (int l = 0; l < 6; ++l) {
    hier_layer<<<LAY[l], 256, 0, stream>>>(
        zg, hout,
        (const float*)d_in[15 + l], (const float*)d_in[21 + l],
        (const float*)d_in[27 + l], (const float*)d_in[33 + l],
        (const float*)d_in[39 + l], (const float*)d_in[45 + l],
        (const float*)d_in[51 + l], (const float*)d_in[57 + l],
        out, off[l], l ? off[l - 1] : 0, l ? 24 : 0);
  }

  final_head<<<(BATCH + 255) / 256, 256, 0, stream>>>(hout, Wfin, bfin, out, off[5]);
}
